// my_model_71897752535695
// MI455X (gfx1250) — matrix operand order repaired and checked by reading, not compiled, not run on hardware
//
#include <hip/hip_runtime.h>
#include <hip/hip_bf16.h>

#define NA_N 50000
#define NB_N 50000
#define NPAD 50016   // rows padded to a multiple of 32 for 2x4-tile GEMM blocking
#define NE   800000
#define NGR  512

typedef __bf16 bf16_t;
typedef __attribute__((ext_vector_type(16))) __bf16 v16bf;
typedef __attribute__((ext_vector_type(8)))  float  v8f;
typedef __attribute__((ext_vector_type(4)))  unsigned int u32x4;

union V16U { u32x4 u[2]; v16bf v; };

static inline unsigned cdiv(long long a, long long b) { return (unsigned)((a + b - 1) / b); }

// ---------------- utility kernels ----------------

__global__ void k_zero(float* __restrict__ p, long long n) {
  long long i = (long long)blockIdx.x * blockDim.x + threadIdx.x;
  long long st = (long long)gridDim.x * blockDim.x;
  for (; i < n; i += st) p[i] = 0.0f;
}

// histogram of indices (edge in-degree / per-graph node counts)
__global__ void k_count(const int* __restrict__ idx, float* __restrict__ cnt, int n) {
  int i = blockIdx.x * blockDim.x + threadIdx.x;
  if (i < n) unsafeAtomicAdd(&cnt[idx[i]], 1.0f);
}

__global__ void k_inv(float* __restrict__ c, int n) {
  int i = blockIdx.x * blockDim.x + threadIdx.x;
  if (i < n) c[i] = 1.0f / fmaxf(c[i], 1.0f);
}

// ---------------- edge scatter (segment_sum of gathered features) ----------------
// one thread = one float4 chunk of one edge message; HW f32 atomics into agg (L2-resident)
__global__ void k_scatter(const int* __restrict__ src, const int* __restrict__ dst,
                          const float* __restrict__ x, float* __restrict__ agg,
                          int F, int ne) {
  const int q = F >> 2;
  long long tid = (long long)blockIdx.x * blockDim.x + threadIdx.x;
  long long total = (long long)ne * q;
  if (tid >= total) return;
  int e = (int)(tid / q);
  int c = (int)(tid - (long long)e * q) << 2;
  int s = src[e], d = dst[e];
  const float4 v = *(const float4*)(x + (long long)s * F + c);
  float* ap = agg + (long long)d * F + c;
  unsafeAtomicAdd(ap + 0, v.x);
  unsafeAtomicAdd(ap + 1, v.y);
  unsafeAtomicAdd(ap + 2, v.z);
  unsafeAtomicAdd(ap + 3, v.w);
}

// ---------------- build concatenated bf16 operands ----------------
// A_cat[n, 0:F) = agg[n]*invdeg[n] (mean);  A_cat[n, F:2F) = h[n] (self)
__global__ void k_acat(const float* __restrict__ agg, const float* __restrict__ inv,
                       const float* __restrict__ h, bf16_t* __restrict__ acat,
                       int N, int F) {
  long long tid = (long long)blockIdx.x * blockDim.x + threadIdx.x;
  long long total = (long long)N * F;
  if (tid >= total) return;
  int n = (int)(tid / F);
  int f = (int)(tid - (long long)n * F);
  float ic = inv[n];
  acat[(long long)n * 2 * F + f]     = (bf16_t)(agg[(long long)n * F + f] * ic);
  acat[(long long)n * 2 * F + F + f] = (bf16_t)(h[(long long)n * F + f]);
}

// B_cat stored transposed: bcat[o, k] with k<din -> Wl[k,o], else Wr[k-din,o]
__global__ void k_bcat(const float* __restrict__ Wl, const float* __restrict__ Wr,
                       bf16_t* __restrict__ bcat, int din, int dout) {
  int K = 2 * din;
  int tid = blockIdx.x * blockDim.x + threadIdx.x;
  if (tid >= dout * K) return;
  int o = tid / K;
  int k = tid - o * K;
  float v = (k < din) ? Wl[k * dout + o] : Wr[(k - din) * dout + o];
  bcat[(long long)o * K + k] = (bf16_t)v;
}

// ---------------- WMMA GEMM: out[NPAD,dout] = A[NPAD,K] * B^T[dout,K] + bias (+relu) ----
// one wave32 computes a 32x64 C block (2 M-tiles x 4 N-tiles) -> 8 WMMAs per K-step,
// each A fragment reused 4x, each B fragment reused 2x.
__global__ void k_gemm(const bf16_t* __restrict__ A, const bf16_t* __restrict__ B,
                       const float* __restrict__ bias, float* __restrict__ out,
                       int K, int dout, int relu, int ngroups) {
  int wave = threadIdx.x >> 5;
  int lane = threadIdx.x & 31;
  int g = blockIdx.x * (blockDim.x >> 5) + wave;
  if (g >= ngroups) return;                // wave-uniform: EXEC stays all-ones for WMMA
  int NG4 = dout >> 6;                     // N-groups of 64 columns
  int mg = g / NG4, ng = g - mg * NG4;
  int m0 = mg << 5;                        // 32 rows
  int n0 = ng << 6;                        // 64 cols
  int half = lane >> 4, lr = lane & 15;

  const bf16_t* arow0 = A + (long long)(m0 + lr) * K;
  const bf16_t* arow1 = arow0 + (long long)16 * K;
  const bf16_t* brow0 = B + (long long)(n0 + lr) * K;
  const bf16_t* brow1 = brow0 + (long long)16 * K;
  const bf16_t* brow2 = brow0 + (long long)32 * K;
  const bf16_t* brow3 = brow0 + (long long)48 * K;

  v8f acc[2][4] = {};
  for (int k0 = 0; k0 < K; k0 += 32) {
    V16U a0, a1, b0, b1, b2, b3;
    // A 16x32 bf16 fragment: lanes 0-15 hold K=[0..15]; lanes 16-31 K=[16..31]
    a0.u[0] = *(const u32x4*)(arow0 + k0 + half * 16);
    a0.u[1] = *(const u32x4*)(arow0 + k0 + half * 16 + 8);
    a1.u[0] = *(const u32x4*)(arow1 + k0 + half * 16);
    a1.u[1] = *(const u32x4*)(arow1 + k0 + half * 16 + 8);
    // B 32x16 bf16 fragment: lane = column; lanes 0-15 hold K=[0..15]; lanes 16-31 K=[16..31]
    b0.u[0] = *(const u32x4*)(brow0 + k0 + half * 16);
    b0.u[1] = *(const u32x4*)(brow0 + k0 + half * 16 + 8);
    b1.u[0] = *(const u32x4*)(brow1 + k0 + half * 16);
    b1.u[1] = *(const u32x4*)(brow1 + k0 + half * 16 + 8);
    b2.u[0] = *(const u32x4*)(brow2 + k0 + half * 16);
    b2.u[1] = *(const u32x4*)(brow2 + k0 + half * 16 + 8);
    b3.u[0] = *(const u32x4*)(brow3 + k0 + half * 16);
    b3.u[1] = *(const u32x4*)(brow3 + k0 + half * 16 + 8);

    acc[0][0] = __builtin_amdgcn_wmma_f32_16x16x32_bf16(false, a0.v, false, b0.v, (short)0, acc[0][0], false, false);
    acc[0][1] = __builtin_amdgcn_wmma_f32_16x16x32_bf16(false, a0.v, false, b1.v, (short)0, acc[0][1], false, false);
    acc[0][2] = __builtin_amdgcn_wmma_f32_16x16x32_bf16(false, a0.v, false, b2.v, (short)0, acc[0][2], false, false);
    acc[0][3] = __builtin_amdgcn_wmma_f32_16x16x32_bf16(false, a0.v, false, b3.v, (short)0, acc[0][3], false, false);
    acc[1][0] = __builtin_amdgcn_wmma_f32_16x16x32_bf16(false, a1.v, false, b0.v, (short)0, acc[1][0], false, false);
    acc[1][1] = __builtin_amdgcn_wmma_f32_16x16x32_bf16(false, a1.v, false, b1.v, (short)0, acc[1][1], false, false);
    acc[1][2] = __builtin_amdgcn_wmma_f32_16x16x32_bf16(false, a1.v, false, b2.v, (short)0, acc[1][2], false, false);
    acc[1][3] = __builtin_amdgcn_wmma_f32_16x16x32_bf16(false, a1.v, false, b3.v, (short)0, acc[1][3], false, false);
  }

#pragma unroll
  for (int i = 0; i < 2; ++i) {
#pragma unroll
    for (int j = 0; j < 4; ++j) {
      int col = n0 + 16 * j + lr;
      float bv = bias[col];
#pragma unroll
      for (int r = 0; r < 8; ++r) {
        // C/D layout: VGPR r -> M = r + 8*half, N = lane&15
        float v = acc[i][j][r] + bv;
        if (relu) v = fmaxf(v, 0.0f);
        out[(long long)(m0 + 16 * i + r + 8 * half) * dout + col] = v;
      }
    }
  }
}

// ---------------- graph mean-pool ----------------
__global__ void k_pool(const float* __restrict__ h, const int* __restrict__ batch,
                       float* __restrict__ pooled, int n) {
  long long tid = (long long)blockIdx.x * blockDim.x + threadIdx.x;
  if (tid >= (long long)n * 128) return;
  int nd = (int)(tid >> 7);
  int f = (int)(tid & 127);
  unsafeAtomicAdd(&pooled[(long long)batch[nd] * 128 + f], h[(long long)nd * 128 + f]);
}

// out[g, j] = sum_f (pa[g,f]/ca + pb[g,f]/cb) * Wlin[f,j] + blin[j]
__global__ void k_final(const float* __restrict__ pa, const float* __restrict__ pb,
                        const float* __restrict__ gca, const float* __restrict__ gcb,
                        const float* __restrict__ Wlin, const float* __restrict__ blin,
                        float* __restrict__ out) {
  int tid = blockIdx.x * blockDim.x + threadIdx.x;
  if (tid >= NGR * 2) return;
  int g = tid >> 1, j = tid & 1;
  float ia = 1.0f / fmaxf(gca[g], 1.0f);
  float ib = 1.0f / fmaxf(gcb[g], 1.0f);
  float s = blin[j];
  for (int f = 0; f < 128; ++f) {
    float p = pa[g * 128 + f] * ia + pb[g * 128 + f] * ib;
    s += p * Wlin[f * 2 + j];
  }
  out[g * 2 + j] = s;
}

// ---------------- host orchestration ----------------

extern "C" void kernel_launch(void* const* d_in, const int* in_sizes, int n_in,
                              void* d_out, int out_size, void* d_ws, size_t ws_size,
                              hipStream_t stream) {
  (void)in_sizes; (void)n_in; (void)out_size; (void)ws_size;

  const float* x_a = (const float*)d_in[0];
  const float* x_b = (const float*)d_in[1];
  const float* P[20];
  for (int i = 0; i < 20; ++i) P[i] = (const float*)d_in[2 + i];
  // params order per layer l: [Wl_ab, bl_ab, Wr_ab, Wl_ba, bl_ba, Wr_ba]; then W_lin, b_lin
  const int* e_ab    = (const int*)d_in[22];  // row0: src in a, row1: dst in b
  const int* e_ba    = (const int*)d_in[23];  // row0: src in b, row1: dst in a
  const int* batch_a = (const int*)d_in[24];
  const int* batch_b = (const int*)d_in[25];

  char* w = (char*)d_ws;
  auto carve = [&](size_t bytes) -> void* {
    void* p = (void*)w;
    w += (bytes + 255) & ~(size_t)255;
    return p;
  };
  float*  hA[2]; float* hB[2];
  hA[0] = (float*)carve((size_t)NPAD * 256 * 4);
  hA[1] = (float*)carve((size_t)NPAD * 256 * 4);
  hB[0] = (float*)carve((size_t)NPAD * 256 * 4);
  hB[1] = (float*)carve((size_t)NPAD * 256 * 4);
  float*  agg      = (float*)carve((size_t)NA_N * 256 * 4);
  bf16_t* acat     = (bf16_t*)carve((size_t)NPAD * 512 * 2);
  bf16_t* bcat     = (bf16_t*)carve((size_t)256 * 512 * 2);
  float*  icnt_a   = (float*)carve((size_t)NA_N * 4);
  float*  icnt_b   = (float*)carve((size_t)NB_N * 4);
  float*  pooled_a = (float*)carve((size_t)NGR * 128 * 4);
  float*  pooled_b = (float*)carve((size_t)NGR * 128 * 4);
  float*  gcnt_a   = (float*)carve((size_t)NGR * 4);
  float*  gcnt_b   = (float*)carve((size_t)NGR * 4);

  const int B = 256;

  // in-degree reciprocals (fixed across layers)
  k_zero<<<cdiv(NA_N, B), B, 0, stream>>>(icnt_a, NA_N);
  k_zero<<<cdiv(NB_N, B), B, 0, stream>>>(icnt_b, NB_N);
  k_count<<<cdiv(NE, B), B, 0, stream>>>(e_ba + NE, icnt_a, NE);
  k_count<<<cdiv(NE, B), B, 0, stream>>>(e_ab + NE, icnt_b, NE);
  k_inv<<<cdiv(NA_N, B), B, 0, stream>>>(icnt_a, NA_N);
  k_inv<<<cdiv(NB_N, B), B, 0, stream>>>(icnt_b, NB_N);

  const int din[3] = {128, 256, 256};
  const int dou[3] = {256, 256, 128};
  const float* cur_a = x_a;
  const float* cur_b = x_b;

  for (int l = 0; l < 3; ++l) {
    const int F = din[l], D = dou[l], K = 2 * F;
    const int relu = (l < 2);
    float* out_a = hA[l & 1];
    float* out_b = hB[l & 1];
    const float *Wl_ab = P[l * 6 + 0], *bl_ab = P[l * 6 + 1], *Wr_ab = P[l * 6 + 2];
    const float *Wl_ba = P[l * 6 + 3], *bl_ba = P[l * 6 + 4], *Wr_ba = P[l * 6 + 5];
    const long long aggA = (long long)NA_N * F;
    const long long aggB = (long long)NB_N * F;
    const long long sctr = (long long)NE * (F >> 2);
    const int ngroups = (NPAD / 32) * (D / 64);  // 32x64 C block per wave

    // ---- update type-a nodes (edges b -> a)
    k_zero<<<cdiv(aggA, B), B, 0, stream>>>(agg, aggA);
    k_scatter<<<cdiv(sctr, B), B, 0, stream>>>(e_ba, e_ba + NE, cur_b, agg, F, NE);
    k_acat<<<cdiv(aggA, B), B, 0, stream>>>(agg, icnt_a, cur_a, acat, NA_N, F);
    k_bcat<<<cdiv(D * K, B), B, 0, stream>>>(Wl_ba, Wr_ba, bcat, F, D);
    k_gemm<<<cdiv(ngroups, 8), 256, 0, stream>>>(acat, bcat, bl_ba, out_a, K, D, relu, ngroups);

    // ---- update type-b nodes (edges a -> b)
    k_zero<<<cdiv(aggB, B), B, 0, stream>>>(agg, aggB);
    k_scatter<<<cdiv(sctr, B), B, 0, stream>>>(e_ab, e_ab + NE, cur_a, agg, F, NE);
    k_acat<<<cdiv(aggB, B), B, 0, stream>>>(agg, icnt_b, cur_b, acat, NB_N, F);
    k_bcat<<<cdiv(D * K, B), B, 0, stream>>>(Wl_ab, Wr_ab, bcat, F, D);
    k_gemm<<<cdiv(ngroups, 8), 256, 0, stream>>>(acat, bcat, bl_ab, out_b, K, D, relu, ngroups);

    cur_a = out_a;
    cur_b = out_b;
  }

  // ---- graph mean-pool (final dim = 128) + head
  k_zero<<<cdiv(NGR * 128, B), B, 0, stream>>>(pooled_a, NGR * 128);
  k_zero<<<cdiv(NGR * 128, B), B, 0, stream>>>(pooled_b, NGR * 128);
  k_zero<<<cdiv(NGR, B), B, 0, stream>>>(gcnt_a, NGR);
  k_zero<<<cdiv(NGR, B), B, 0, stream>>>(gcnt_b, NGR);
  k_count<<<cdiv(NA_N, B), B, 0, stream>>>(batch_a, gcnt_a, NA_N);
  k_count<<<cdiv(NB_N, B), B, 0, stream>>>(batch_b, gcnt_b, NB_N);
  k_pool<<<cdiv((long long)NA_N * 128, B), B, 0, stream>>>(cur_a, batch_a, pooled_a, NA_N);
  k_pool<<<cdiv((long long)NB_N * 128, B), B, 0, stream>>>(cur_b, batch_b, pooled_b, NB_N);
  k_final<<<cdiv(NGR * 2, 256), 256, 0, stream>>>(pooled_a, pooled_b, gcnt_a, gcnt_b,
                                                  P[18], P[19], (float*)d_out);
}
